// SWAStatefulChunk1Prefill_90194313216376
// MI455X (gfx1250) — matrix operand order repaired and checked by reading, not compiled, not run on hardware
//
#include <hip/hip_runtime.h>
#include <hip/hip_bf16.h>

// ---------------------------------------------------------------------------
// Model dims (from the reference)
// ---------------------------------------------------------------------------
#define H_   8
#define KV_  2
#define HD_  256
#define CTX_ 4096
#define T_   1024
#define D_   2048
#define FF_  8192
#define PLD_ 256
#define NREP (H_ / KV_)

typedef __attribute__((ext_vector_type(16))) __bf16 v16bf;
typedef __attribute__((ext_vector_type(8)))  float  v8f;
typedef __attribute__((ext_vector_type(4)))  int    v4i;

static __device__ __forceinline__ __bf16 tobf(float f) { return (__bf16)f; }

static __device__ __forceinline__ float gelu_tanh(float x) {
    float t = 0.7978845608028654f * (x + 0.044715f * x * x * x);
    return 0.5f * x * (1.0f + tanhf(t));
}

// ---- CDNA5 async LDS copy (guarded; falls back to sync copy if absent) ----
#if __has_builtin(__builtin_amdgcn_global_load_async_to_lds_b128)
#define USE_ASYNC_LDS 1
#else
#define USE_ASYNC_LDS 0
#endif

#if USE_ASYNC_LDS
static __device__ __forceinline__ void async_copy_b128(const void* gsrc, void* ldst) {
    // Global flat addresses == AS1 addresses; LDS aperture maps addr[31:0]
    // directly to the LDS offset, so integer casts are address-correct.
    __builtin_amdgcn_global_load_async_to_lds_b128(
        (__attribute__((address_space(1))) v4i*)(uintptr_t)gsrc,
        (__attribute__((address_space(3))) v4i*)(uintptr_t)ldst,
        0, 0);
}
static __device__ __forceinline__ void wait_async0() {
#if __has_builtin(__builtin_amdgcn_s_wait_asynccnt)
    __builtin_amdgcn_s_wait_asynccnt(0);
#else
    asm volatile("s_wait_asynccnt 0x0" ::: "memory");
#endif
}
#endif

// ---------------------------------------------------------------------------
// Generic row-RMSNorm:  out[row] = (res? res[row]:0) + rms(x[row])*(1+w),
// optionally scaled by *scal.  One block per row.
// ---------------------------------------------------------------------------
__global__ __launch_bounds__(256) void row_rms(
    const float* __restrict__ x, const float* __restrict__ w,
    const float* __restrict__ res, const float* __restrict__ scal,
    float* __restrict__ out, int C)
{
    const int row = blockIdx.x;
    const int tid = threadIdx.x;
    __shared__ float red[256];
    float ss = 0.f;
    for (int c = tid; c < C; c += 256) {
        float v = x[(size_t)row * C + c];
        ss += v * v;
    }
    red[tid] = ss;
    __syncthreads();
    for (int s = 128; s > 0; s >>= 1) {
        if (tid < s) red[tid] += red[tid + s];
        __syncthreads();
    }
    const float scale = rsqrtf(red[0] / (float)C + 1e-6f);
    const float mul = scal ? scal[0] : 1.0f;
    for (int c = tid; c < C; c += 256) {
        float v = x[(size_t)row * C + c] * scale * (w ? (1.0f + w[c]) : 1.0f);
        if (res) v += res[(size_t)row * C + c];
        out[(size_t)row * C + c] = v * mul;
    }
}

// ---------------------------------------------------------------------------
// Per-head RMSNorm (+ optional RoPE) with scatter to
// out[(head*out_tstride + pos + t)*HD + d].  One block (HD threads) per row.
// ---------------------------------------------------------------------------
__global__ __launch_bounds__(HD_) void head_norm_rope(
    const float* __restrict__ x,      // [T][nh*HD]
    const float* __restrict__ w,      // [HD] or null
    const float* __restrict__ cos_p,  // [T][HD] or null
    const float* __restrict__ sin_p,  // [T][HD] or null
    float* __restrict__ out, int nh, int out_tstride,
    const int* __restrict__ posp)
{
    const int row  = blockIdx.x;       // 0 .. nh*T-1
    const int head = row / T_;
    const int t    = row % T_;
    const int d    = threadIdx.x;      // 0..255

    __shared__ float red[HD_];
    __shared__ float sh[HD_];

    float v = x[(size_t)t * nh * HD_ + (size_t)head * HD_ + d];
    red[d] = v * v;
    __syncthreads();
    for (int s = 128; s > 0; s >>= 1) {
        if (d < s) red[d] += red[d + s];
        __syncthreads();
    }
    const float scale = rsqrtf(red[0] / (float)HD_ + 1e-6f);
    float y = v * scale * (w ? (1.0f + w[d]) : 1.0f);
    float o = y;
    if (cos_p) {
        sh[d] = y;
        __syncthreads();
        const float rot = (d < HD_ / 2) ? -sh[d + HD_ / 2] : sh[d - HD_ / 2];
        o = y * cos_p[(size_t)t * HD_ + d] + rot * sin_p[(size_t)t * HD_ + d];
    }
    const int pos = posp ? posp[0] : 0;
    out[((size_t)head * out_tstride + pos + t) * HD_ + d] = o;
}

// ---------------------------------------------------------------------------
// GELU(g) * u elementwise
// ---------------------------------------------------------------------------
__global__ __launch_bounds__(256) void geglu_mul(
    const float* __restrict__ g, const float* __restrict__ u,
    float* __restrict__ o, int n)
{
    int i = blockIdx.x * 256 + threadIdx.x;
    if (i < n) o[i] = gelu_tanh(g[i]) * u[i];
}

// ---------------------------------------------------------------------------
// f32 -> bf16 bulk convert (used once per launch on the updated KV cache)
// ---------------------------------------------------------------------------
__global__ __launch_bounds__(256) void cvt_f32_to_bf16(
    const float* __restrict__ in, __bf16* __restrict__ out, int n)
{
    int i = blockIdx.x * 256 + threadIdx.x;
    if (i < n) out[i] = tobf(in[i]);
}

// ---------------------------------------------------------------------------
// bf16 WMMA GEMM:  C[M,N] = A[M,K] @ B[K,N]   (f32 in memory, bf16 in LDS,
// f32 accumulation).  BM=128, BN=128, BK=32.  256 threads = 8 waves (2x4);
// each wave owns a 64x32 output tile = 4x2 WMMA accumulators.
// ---------------------------------------------------------------------------
#define BM 128
#define BN 128
#define BK 32

__global__ __launch_bounds__(256) void gemm_bf16_wmma(
    const float* __restrict__ A, const float* __restrict__ B,
    float* __restrict__ C, int M, int N, int K)
{
    __shared__ __bf16 Asb[BM][BK];   // row-major (A-fragment friendly)
    __shared__ __bf16 Bsb[BN][BK];   // B stored transposed: [n][k]

    const int tid  = threadIdx.x;
    const int lane = tid & 31;
    const int wid  = tid >> 5;
    const int hsel = lane >> 4;
    const int l16  = lane & 15;
    const int bM = blockIdx.y * BM;
    const int bN = blockIdx.x * BN;
    const int rowBase = (wid >> 2) * 64;
    const int colBase = (wid & 3) * 32;

    v8f acc[4][2] = {};

    for (int k0 = 0; k0 < K; k0 += BK) {
        // ---- issue ALL tile loads first (8 B128 loads in flight) ----
        float4 av[4], bv[4];
        #pragma unroll
        for (int it = 0; it < 4; ++it) {
            int f  = tid + it * 256;
            int r  = f >> 3;
            int c4 = (f & 7) << 2;
            av[it] = *(const float4*)(A + (size_t)(bM + r) * K + k0 + c4);
        }
        #pragma unroll
        for (int it = 0; it < 4; ++it) {
            int f  = tid + it * 256;
            int kk = f >> 5;
            int n4 = (f & 31) << 2;
            bv[it] = *(const float4*)(B + (size_t)(k0 + kk) * N + bN + n4);
        }
        // ---- convert + store to LDS ----
        #pragma unroll
        for (int it = 0; it < 4; ++it) {
            int f  = tid + it * 256;
            int r  = f >> 3;
            int c4 = (f & 7) << 2;
            Asb[r][c4 + 0] = tobf(av[it].x);
            Asb[r][c4 + 1] = tobf(av[it].y);
            Asb[r][c4 + 2] = tobf(av[it].z);
            Asb[r][c4 + 3] = tobf(av[it].w);
        }
        #pragma unroll
        for (int it = 0; it < 4; ++it) {
            int f  = tid + it * 256;
            int kk = f >> 5;
            int n4 = (f & 31) << 2;
            Bsb[n4 + 0][kk] = tobf(bv[it].x);
            Bsb[n4 + 1][kk] = tobf(bv[it].y);
            Bsb[n4 + 2][kk] = tobf(bv[it].z);
            Bsb[n4 + 3][kk] = tobf(bv[it].w);
        }
        if (k0 + BK < K) {
            __builtin_prefetch(A + (size_t)(bM + (tid >> 1)) * K + k0 + BK + (tid & 1) * 16, 0, 3);
            __builtin_prefetch(B + (size_t)(k0 + BK + (tid >> 3)) * N + bN + (tid & 7) * 16, 0, 3);
        }
        __syncthreads();

        // ---- fragments + 8 WMMAs ----
        v16bf aF[4], bF[2];
        #pragma unroll
        for (int m = 0; m < 4; ++m) {
            const __bf16* ap = &Asb[rowBase + m * 16 + l16][hsel * 8];
            #pragma unroll
            for (int i = 0; i < 8; ++i) { aF[m][i] = ap[i]; aF[m][8 + i] = ap[16 + i]; }
        }
        #pragma unroll
        for (int n = 0; n < 2; ++n) {
            const __bf16* bp = &Bsb[colBase + n * 16 + l16][hsel * 8];
            #pragma unroll
            for (int i = 0; i < 8; ++i) { bF[n][i] = bp[i]; bF[n][8 + i] = bp[16 + i]; }
        }
        #pragma unroll
        for (int m = 0; m < 4; ++m)
            #pragma unroll
            for (int n = 0; n < 2; ++n)
                acc[m][n] = __builtin_amdgcn_wmma_f32_16x16x32_bf16(
                    false, aF[m], false, bF[n], (short)0, acc[m][n], false, false);
        __syncthreads();
    }

    #pragma unroll
    for (int m = 0; m < 4; ++m)
        #pragma unroll
        for (int n = 0; n < 2; ++n)
            #pragma unroll
            for (int r = 0; r < 8; ++r)
                C[(size_t)(bM + rowBase + m * 16 + r + 8 * hsel) * N
                  + (bN + colBase + n * 16 + l16)] = acc[m][n][r];
}

// ---------------------------------------------------------------------------
// Flash attention with WMMA over a bf16 KV cache.
// One block = (head, 64 query rows), 4 waves, 16 query rows per wave.
// Chunk of 32 keys:  S = q@K^T (16 WMMAs), online softmax, O += P@V (16 WMMAs).
// K tile staged with GLOBAL_LOAD_ASYNC_TO_LDS when available.
// ---------------------------------------------------------------------------
__global__ __launch_bounds__(128) void flash_attn_wmma(
    const float*  __restrict__ qf,   // [H][T][HD] f32
    const __bf16* __restrict__ kbf,  // [KV][CTX][HD] bf16
    const __bf16* __restrict__ vbf,  // [KV][CTX][HD] bf16
    const float*  __restrict__ mask, // [T][CTX]
    float* __restrict__ out)         // [T][H*HD]
{
    __shared__ __bf16 Ks[32][HD_];    // keys row-major (B-frag friendly)
    __shared__ __bf16 Vt[HD_][32];    // V transposed: [hd][key]
    __shared__ __bf16 Pl[4][16][32];  // per-wave P tile

    const int tid  = threadIdx.x;
    const int lane = tid & 31;
    const int w    = tid >> 5;
    const int hsel = lane >> 4;
    const int l16  = lane & 15;
    const int h    = blockIdx.x;
    const int qb   = blockIdx.y * 64;
    const int g    = h / NREP;
    const int tq   = qb + w * 16;

    // ---- load q fragments once (16 rows x 256 dims = 8 A-frags) ----
    v16bf qF[8];
    #pragma unroll
    for (int f = 0; f < 8; ++f) {
        const float* qp = qf + ((size_t)h * T_ + tq + l16) * HD_ + f * 32;
        #pragma unroll
        for (int i = 0; i < 8; ++i) {
            qF[f][i]     = tobf(qp[hsel * 8 + i]);
            qF[f][8 + i] = tobf(qp[16 + hsel * 8 + i]);
        }
    }

    v8f O[16] = {};
    float mrow[8], lrow[8];
    #pragma unroll
    for (int r = 0; r < 8; ++r) { mrow[r] = -1e30f; lrow[r] = 0.f; }

    for (int s0 = 0; s0 < CTX_; s0 += 32) {
        // ---- stage K chunk (32x128 bf16 = 512 x B128 over 128 threads) ----
        #pragma unroll
        for (int it = 0; it < 4; ++it) {
            int f  = tid + it * 128;          // 0..511
            int kr = f >> 4;                  // key 0..31
            int c8 = (f & 15) << 3;           // dim 0..120 step 8
            const __bf16* gsrc = kbf + ((size_t)g * CTX_ + s0 + kr) * HD_ + c8;
#if USE_ASYNC_LDS
            async_copy_b128(gsrc, &Ks[kr][c8]);
#else
            *(uint4*)&Ks[kr][c8] = *(const uint4*)gsrc;
#endif
        }
        // ---- stage V chunk transposed: unpack 8 bf16 per B128 load ----
        #pragma unroll
        for (int it = 0; it < 4; ++it) {
            int f  = tid + it * 128;
            int kr = f >> 4;
            int c8 = (f & 15) << 3;
            const uint4 vv = *(const uint4*)(vbf + ((size_t)g * CTX_ + s0 + kr) * HD_ + c8);
            unsigned int ww[4] = {vv.x, vv.y, vv.z, vv.w};
            #pragma unroll
            for (int j = 0; j < 4; ++j) {
                *(unsigned short*)&Vt[c8 + 2 * j + 0][kr] = (unsigned short)(ww[j] & 0xffffu);
                *(unsigned short*)&Vt[c8 + 2 * j + 1][kr] = (unsigned short)(ww[j] >> 16);
            }
        }
#if USE_ASYNC_LDS
        wait_async0();
#endif
        __syncthreads();

        // ---- S = q @ K^T for the two 16-key tiles ----
        v8f S0 = {}, S1 = {};
        #pragma unroll
        for (int f = 0; f < 8; ++f) {
            v16bf b0, b1;
            const __bf16* bp0 = &Ks[l16][f * 32 + hsel * 8];
            const __bf16* bp1 = &Ks[16 + l16][f * 32 + hsel * 8];
            #pragma unroll
            for (int i = 0; i < 8; ++i) { b0[i] = bp0[i]; b1[i] = bp1[i]; b0[8 + i] = bp0[16 + i]; b1[8 + i] = bp1[16 + i]; }
            S0 = __builtin_amdgcn_wmma_f32_16x16x32_bf16(false, qF[f], false, b0, (short)0, S0, false, false);
            S1 = __builtin_amdgcn_wmma_f32_16x16x32_bf16(false, qF[f], false, b1, (short)0, S1, false, false);
        }

        // ---- online softmax ----
        #pragma unroll
        for (int r = 0; r < 8; ++r) {
            const int trow = tq + 8 * hsel + r;
            float x0 = S0[r] + mask[(size_t)trow * CTX_ + s0 + l16];
            float x1 = S1[r] + mask[(size_t)trow * CTX_ + s0 + 16 + l16];
            float cm = fmaxf(x0, x1);
            #pragma unroll
            for (int d = 1; d < 16; d <<= 1) cm = fmaxf(cm, __shfl_xor(cm, d));
            const float mn = fmaxf(mrow[r], cm);
            const float sc = __expf(mrow[r] - mn);
            const float p0 = __expf(x0 - mn);
            const float p1 = __expf(x1 - mn);
            float rs = p0 + p1;
            #pragma unroll
            for (int d = 1; d < 16; d <<= 1) rs += __shfl_xor(rs, d);
            lrow[r] = lrow[r] * sc + rs;
            mrow[r] = mn;
            #pragma unroll
            for (int j = 0; j < 16; ++j) O[j][r] *= sc;
            Pl[w][r + 8 * hsel][l16]      = tobf(p0);
            Pl[w][r + 8 * hsel][16 + l16] = tobf(p1);
        }
        asm volatile("" ::: "memory");  // keep DS store->load order (wave-private tile)

        // ---- re-shape P into an A-fragment ----
        v16bf pF;
        {
            const __bf16* pp = &Pl[w][l16][0];
            #pragma unroll
            for (int i = 0; i < 8; ++i) {
                pF[i]     = pp[hsel * 8 + i];
                pF[8 + i] = pp[16 + hsel * 8 + i];
            }
        }
        // ---- O += P @ V (16 hd tiles) ----
        #pragma unroll
        for (int j = 0; j < 16; ++j) {
            v16bf vF;
            const __bf16* vp = &Vt[j * 16 + l16][hsel * 8];
            #pragma unroll
            for (int i = 0; i < 8; ++i) { vF[i] = vp[i]; vF[8 + i] = vp[16 + i]; }
            O[j] = __builtin_amdgcn_wmma_f32_16x16x32_bf16(false, pF, false, vF, (short)0, O[j], false, false);
        }
        __syncthreads();
    }

    // ---- epilogue: O /= l, scatter to [T][H*HD] ----
    #pragma unroll
    for (int r = 0; r < 8; ++r) {
        const float inv = 1.0f / lrow[r];
        const int t = tq + 8 * hsel + r;
        #pragma unroll
        for (int j = 0; j < 16; ++j)
            out[(size_t)t * (H_ * HD_) + (size_t)h * HD_ + j * 16 + l16] = O[j][r] * inv;
    }
}

// ---------------------------------------------------------------------------
// Host-side launch
// ---------------------------------------------------------------------------
extern "C" void kernel_launch(void* const* d_in, const int* in_sizes, int n_in,
                              void* d_out, int out_size, void* d_ws, size_t ws_size,
                              hipStream_t stream) {
    (void)in_sizes; (void)n_in; (void)out_size; (void)ws_size;

    const float* hs      = (const float*)d_in[0];
    const float* cos_f   = (const float*)d_in[1];
    const float* sin_f   = (const float*)d_in[2];
    const float* maskp   = (const float*)d_in[3];
    const float* kv_in   = (const float*)d_in[4];
    const float* plc     = (const float*)d_in[5];
    const float* wq      = (const float*)d_in[6];
    const float* wk      = (const float*)d_in[7];
    const float* wv      = (const float*)d_in[8];
    const float* wo      = (const float*)d_in[9];
    const float* ln_in   = (const float*)d_in[10];
    const float* qnw     = (const float*)d_in[11];
    const float* knw     = (const float*)d_in[12];
    const float* ln_pa   = (const float*)d_in[13];
    const float* ln_pf   = (const float*)d_in[14];
    const float* wg      = (const float*)d_in[15];
    const float* wu      = (const float*)d_in[16];
    const float* wd      = (const float*)d_in[17];
    const float* ln_pff  = (const float*)d_in[18];
    const float* w_plg   = (const float*)d_in[19];
    const float* w_plp   = (const float*)d_in[20];
    const float* ln_ppl  = (const float*)d_in[21];
    const float* lscal   = (const float*)d_in[22];
    const int*   posp    = (const int*)d_in[23];

    float* out_h  = (float*)d_out;                       // [T][D]
    float* kv_out = out_h + (size_t)T_ * D_;             // [2][KV][CTX][HD]
    float* kc     = kv_out;
    float* vc     = kv_out + (size_t)KV_ * CTX_ * HD_;

    // workspace layout (floats), with lifetime-based reuse
    float* ws   = (float*)d_ws;
    const size_t M1 = 1u << 20;                  // 1M floats
    float* buf0 = ws;                            // 2M: h_norm -> hf -> hidden2
    float* buf1 = ws + 2 * M1;                   // 2M: q_raw -> attn_ctx
    float* buf2 = ws + 4 * M1;                   // 0.5M: k_raw -> pg
    float* buf3 = ws + 4 * M1 + (M1 / 2);        // 0.5M: v_raw -> gated
    float* buf4 = ws + 5 * M1;                   // 2M: q_fin -> plo
    float* buf5 = ws + 7 * M1;                   // 2M: attn_proj -> mlp
    float* buf6 = ws + 9 * M1;                   // 2M: hidden1
    float* buf7 = ws + 11 * M1;                  // 8M: gate
    float* buf8 = ws + 19 * M1;                  // 8M: up
    __bf16* kvbf = (__bf16*)(ws + 27 * M1);      // 4M bf16: KV cache mirror
    __bf16* kbf  = kvbf;
    __bf16* vbf  = kvbf + (size_t)KV_ * CTX_ * HD_;

    // 0) seed output KV cache with full old state (capture-safe async d2d)
    (void)hipMemcpyAsync(kv_out, kv_in, sizeof(float) * 2 * KV_ * CTX_ * HD_,
                         hipMemcpyDeviceToDevice, stream);

    // 1) input RMSNorm
    row_rms<<<T_, 256, 0, stream>>>(hs, ln_in, nullptr, nullptr, buf0, D_);

    // 2) QKV projections
    gemm_bf16_wmma<<<dim3((H_ * HD_) / BN, T_ / BM), 256, 0, stream>>>(buf0, wq, buf1, T_, H_ * HD_, D_);
    gemm_bf16_wmma<<<dim3((KV_ * HD_) / BN, T_ / BM), 256, 0, stream>>>(buf0, wk, buf2, T_, KV_ * HD_, D_);
    gemm_bf16_wmma<<<dim3((KV_ * HD_) / BN, T_ / BM), 256, 0, stream>>>(buf0, wv, buf3, T_, KV_ * HD_, D_);

    // 3) head norms + RoPE; k/v scatter into the cache at current_pos
    head_norm_rope<<<H_ * T_, HD_, 0, stream>>>(buf1, qnw, cos_f, sin_f, buf4, H_, T_, nullptr);
    head_norm_rope<<<KV_ * T_, HD_, 0, stream>>>(buf2, knw, cos_f, sin_f, kc, KV_, CTX_, posp);
    head_norm_rope<<<KV_ * T_, HD_, 0, stream>>>(buf3, nullptr, nullptr, nullptr, vc, KV_, CTX_, posp);

    // 3b) bf16 mirror of the updated cache for matrix-core attention
    cvt_f32_to_bf16<<<(2 * KV_ * CTX_ * HD_) / 256, 256, 0, stream>>>(
        kv_out, kvbf, 2 * KV_ * CTX_ * HD_);

    // 4) attention (flash, WMMA, async LDS staging)
    flash_attn_wmma<<<dim3(H_, T_ / 64), 128, 0, stream>>>(buf4, kbf, vbf, maskp, buf1);

    // 5) output projection + post-attn RMS + residual
    gemm_bf16_wmma<<<dim3(D_ / BN, T_ / BM), 256, 0, stream>>>(buf1, wo, buf5, T_, D_, H_ * HD_);
    row_rms<<<T_, 256, 0, stream>>>(buf5, ln_pa, hs, nullptr, buf6, D_);

    // 6) FFN
    row_rms<<<T_, 256, 0, stream>>>(buf6, ln_pf, nullptr, nullptr, buf0, D_);
    gemm_bf16_wmma<<<dim3(FF_ / BN, T_ / BM), 256, 0, stream>>>(buf0, wg, buf7, T_, FF_, D_);
    gemm_bf16_wmma<<<dim3(FF_ / BN, T_ / BM), 256, 0, stream>>>(buf0, wu, buf8, T_, FF_, D_);
    geglu_mul<<<(T_ * FF_) / 256, 256, 0, stream>>>(buf7, buf8, buf7, T_ * FF_);
    gemm_bf16_wmma<<<dim3(D_ / BN, T_ / BM), 256, 0, stream>>>(buf7, wd, buf5, T_, D_, FF_);
    row_rms<<<T_, 256, 0, stream>>>(buf5, ln_pff, buf6, nullptr, buf0, D_);

    // 7) per-layer projection branch
    gemm_bf16_wmma<<<dim3(PLD_ / BN, T_ / BM), 256, 0, stream>>>(buf0, w_plg, buf2, T_, PLD_, D_);
    geglu_mul<<<(T_ * PLD_) / 256, 256, 0, stream>>>(buf2, plc, buf3, T_ * PLD_);
    gemm_bf16_wmma<<<dim3(D_ / BN, T_ / BM), 256, 0, stream>>>(buf3, w_plp, buf4, T_, D_, PLD_);

    // 8) final RMS + residual + layer scalar -> d_out hidden
    row_rms<<<T_, 256, 0, stream>>>(buf4, ln_ppl, buf0, lscal, out_h, D_);
}
